// LatentODEEncoder_32650341384388
// MI455X (gfx1250) — matrix operand order repaired and checked by reading, not compiled, not run on hardware
//
#include <hip/hip_runtime.h>
#include <hip/hip_bf16.h>
#include <math.h>

// ---------------------------------------------------------------------------
// Types for CDNA5 WMMA
// ---------------------------------------------------------------------------
typedef __attribute__((ext_vector_type(16))) __bf16 v16bf;
typedef __attribute__((ext_vector_type(8)))  float  v8f;

#define ASTRIDE 776   // bf16 elements per row of A buffer (512 h + 256 x + pad)
#define KSTRIDE 520   // bf16 elements per row of k-stage / z buffers
#define YSTRIDE 516   // f32 elements per row of y (ODE state)

// DOPRI5 coefficients
__device__ __constant__ float DA[6][5] = {
  {0.f,0.f,0.f,0.f,0.f},
  {0.2f,0.f,0.f,0.f,0.f},
  {3.f/40.f, 9.f/40.f, 0.f,0.f,0.f},
  {44.f/45.f, -56.f/15.f, 32.f/9.f, 0.f,0.f},
  {19372.f/6561.f, -25360.f/2187.f, 64448.f/6561.f, -212.f/729.f, 0.f},
  {9017.f/3168.f, -355.f/33.f, 46732.f/5247.f, 49.f/176.f, -5103.f/18656.f}
};
__device__ __constant__ float DBC[6] = {
  35.f/384.f, 0.f, 500.f/1113.f, 125.f/192.f, -2187.f/6784.f, 11.f/84.f
};

union FragU { int4 q[2]; v16bf v; };

__device__ __forceinline__ v8f wmma_bf16(v16bf a, v16bf b, v8f c) {
  return __builtin_amdgcn_wmma_f32_16x16x32_bf16(false, a, false, b, (short)0, c,
                                                 false, false);
}

__device__ __forceinline__ v8f zero8() {
  v8f z = {0.f,0.f,0.f,0.f,0.f,0.f,0.f,0.f};
  return z;
}

// Hardware transcendental tanh (CDNA5 TRANS32 op). v_nop inside the asm
// satisfies the "1 independent op after TRANS before result use" hazard.
__device__ __forceinline__ float tanh_hw(float x) {
  float r;
  asm("v_tanh_f32 %0, %1\n\tv_nop" : "=v"(r) : "v"(x));
  return r;
}
__device__ __forceinline__ float sigmoid_hw(float x) {
  return 0.5f + 0.5f * tanh_hw(0.5f * x);
}

// ---------------------------------------------------------------------------
// Fused 4-tile GEMM: one wave computes 4 adjacent 16x16 N-tiles that share the
// same A fragments. Per kf: 1 A load -> 4 B loads -> 4 WMMAs into 4
// independent accumulator chains.
// ---------------------------------------------------------------------------
template <int NKF>
__device__ __forceinline__ void gemm4(const __bf16* Abuf, int astride, int acol0,
                                      const __bf16* Wf, int NT0, int lane,
                                      v8f acc[4]) {
  const int M  = lane & 15;
  const int kh = (lane >> 4) << 4;
  const __bf16* arow = Abuf + M * astride + acol0 + kh;
  const __bf16* wp   = Wf + (size_t)(NT0 * NKF) * 512 + lane * 16;
  acc[0] = zero8(); acc[1] = zero8(); acc[2] = zero8(); acc[3] = zero8();
  #pragma unroll 2
  for (int kf = 0; kf < NKF; ++kf) {
    FragU fa;
    fa.q[0] = *reinterpret_cast<const int4*>(arow + kf * 32);
    fa.q[1] = *reinterpret_cast<const int4*>(arow + kf * 32 + 8);
    #pragma unroll
    for (int nt = 0; nt < 4; ++nt) {
      FragU fb;
      fb.q[0] = *reinterpret_cast<const int4*>(wp + ((size_t)nt * NKF + kf) * 512);
      fb.q[1] = *reinterpret_cast<const int4*>(wp + ((size_t)nt * NKF + kf) * 512 + 8);
      acc[nt] = wmma_bf16(fa.v, fb.v, acc[nt]);
    }
  }
}

// ---------------------------------------------------------------------------
// Fused GRU gate GEMMs for one N-tile: r/z/h_n share A over K=[0,512) (h),
// r/z/i_n share A over K=[512,768) (x). 1 A load -> 3 B loads -> 3 WMMAs.
// ---------------------------------------------------------------------------
__device__ __forceinline__ void gru_gates(
    const __bf16* Abuf, const __bf16* Wrf, const __bf16* Wzf,
    const __bf16* Whnf, const __bf16* Winf, int NT, int lane,
    v8f& accR, v8f& accZ, v8f& accH, v8f& accI) {
  const int M  = lane & 15;
  const int kh = (lane >> 4) << 4;
  const __bf16* arow = Abuf + M * ASTRIDE + kh;
  const __bf16* wr = Wrf  + (size_t)(NT * 24) * 512 + lane * 16;
  const __bf16* wz = Wzf  + (size_t)(NT * 24) * 512 + lane * 16;
  const __bf16* wh = Whnf + (size_t)(NT * 16) * 512 + lane * 16;
  const __bf16* wi = Winf + (size_t)(NT * 8)  * 512 + lane * 16;
  accR = zero8(); accZ = zero8(); accH = zero8(); accI = zero8();
  #pragma unroll 2
  for (int kf = 0; kf < 16; ++kf) {       // h region: r, z, h_n
    FragU fa, f0, f1, f2;
    fa.q[0] = *reinterpret_cast<const int4*>(arow + kf * 32);
    fa.q[1] = *reinterpret_cast<const int4*>(arow + kf * 32 + 8);
    f0.q[0] = *reinterpret_cast<const int4*>(wr + (size_t)kf * 512);
    f0.q[1] = *reinterpret_cast<const int4*>(wr + (size_t)kf * 512 + 8);
    f1.q[0] = *reinterpret_cast<const int4*>(wz + (size_t)kf * 512);
    f1.q[1] = *reinterpret_cast<const int4*>(wz + (size_t)kf * 512 + 8);
    f2.q[0] = *reinterpret_cast<const int4*>(wh + (size_t)kf * 512);
    f2.q[1] = *reinterpret_cast<const int4*>(wh + (size_t)kf * 512 + 8);
    accR = wmma_bf16(fa.v, f0.v, accR);
    accZ = wmma_bf16(fa.v, f1.v, accZ);
    accH = wmma_bf16(fa.v, f2.v, accH);
  }
  #pragma unroll 2
  for (int kf = 0; kf < 8; ++kf) {        // x region: r, z, i_n
    FragU fa, f0, f1, f2;
    fa.q[0] = *reinterpret_cast<const int4*>(arow + 512 + kf * 32);
    fa.q[1] = *reinterpret_cast<const int4*>(arow + 512 + kf * 32 + 8);
    f0.q[0] = *reinterpret_cast<const int4*>(wr + (size_t)(16 + kf) * 512);
    f0.q[1] = *reinterpret_cast<const int4*>(wr + (size_t)(16 + kf) * 512 + 8);
    f1.q[0] = *reinterpret_cast<const int4*>(wz + (size_t)(16 + kf) * 512);
    f1.q[1] = *reinterpret_cast<const int4*>(wz + (size_t)(16 + kf) * 512 + 8);
    f2.q[0] = *reinterpret_cast<const int4*>(wi + (size_t)kf * 512);
    f2.q[1] = *reinterpret_cast<const int4*>(wi + (size_t)kf * 512 + 8);
    accR = wmma_bf16(fa.v, f0.v, accR);
    accZ = wmma_bf16(fa.v, f1.v, accZ);
    accI = wmma_bf16(fa.v, f2.v, accI);
  }
}

// ---------------------------------------------------------------------------
// Weight packing kernels (fp32 -> bf16 WMMA B-fragments in workspace).
// B layout (32x16 bf16): element e of lane L: N = L&15, K = kf*32 + 16*(L>=16) + e
// ---------------------------------------------------------------------------
__global__ void pack_plain(const float* __restrict__ W, int ldk, int n0, int nkf,
                           unsigned short* __restrict__ dst_) {
  __bf16* dst = reinterpret_cast<__bf16*>(dst_);
  const int f    = blockIdx.x;
  const int nt   = f / nkf;
  const int kf   = f - nt * nkf;
  const int lane = threadIdx.x;
  const int n    = n0 + nt * 16 + (lane & 15);
  const int kh   = (lane >> 4) << 4;
  __bf16* d = dst + (size_t)f * 512 + lane * 16;
  #pragma unroll
  for (int e = 0; e < 16; ++e) {
    int k = kf * 32 + kh + e;
    d[e] = (__bf16)W[(size_t)n * ldk + k];
  }
}

__global__ void pack_concat(const float* __restrict__ Whh,
                            const float* __restrict__ Wih, int n0,
                            unsigned short* __restrict__ dst_) {
  __bf16* dst = reinterpret_cast<__bf16*>(dst_);
  const int f    = blockIdx.x;
  const int nt   = f / 24;
  const int kf   = f - nt * 24;
  const int lane = threadIdx.x;
  const int n    = n0 + nt * 16 + (lane & 15);
  const int kh   = (lane >> 4) << 4;
  __bf16* d = dst + (size_t)f * 512 + lane * 16;
  #pragma unroll
  for (int e = 0; e < 16; ++e) {
    int k = kf * 32 + kh + e;
    float v = (k < 512) ? Whh[(size_t)n * 512 + k]
                        : Wih[(size_t)n * 256 + (k - 512)];
    d[e] = (__bf16)v;
  }
}

// ---------------------------------------------------------------------------
// Persistent ODE-GRU kernel. One workgroup owns 16 batch rows for the entire
// 32-step sequence. 256 threads = 8 wave32; wave w owns output cols [64w,64w+64).
// ---------------------------------------------------------------------------
__global__ __launch_bounds__(256) void node_persistent(
    const float* __restrict__ t,    const float* __restrict__ x,
    const float* __restrict__ b_ih, const float* __restrict__ b_hh,
    const float* __restrict__ b1,   const float* __restrict__ b2,
    const unsigned short* __restrict__ W1f_,  const unsigned short* __restrict__ W2f_,
    const unsigned short* __restrict__ Wrf_,  const unsigned short* __restrict__ Wzf_,
    const unsigned short* __restrict__ Winf_, const unsigned short* __restrict__ Whnf_,
    float* __restrict__ out)
{
  const __bf16* W1f  = reinterpret_cast<const __bf16*>(W1f_);
  const __bf16* W2f  = reinterpret_cast<const __bf16*>(W2f_);
  const __bf16* Wrf  = reinterpret_cast<const __bf16*>(Wrf_);
  const __bf16* Wzf  = reinterpret_cast<const __bf16*>(Wzf_);
  const __bf16* Winf = reinterpret_cast<const __bf16*>(Winf_);
  const __bf16* Whnf = reinterpret_cast<const __bf16*>(Whnf_);

  __shared__ float  ys[16 * YSTRIDE];      //  33 KB : ODE state y (fp32 master)
  __shared__ __bf16 kb[5][16 * KSTRIDE];   //  83 KB : DOPRI5 stages k1..k5 (bf16)
  __shared__ __bf16 Ab[16 * ASTRIDE];      //  25 KB : WMMA A buffer [h | x]
  __shared__ __bf16 Zb[16 * KSTRIDE];      //  17 KB : relu(y@W1+b1) staging
  __shared__ float  dtl[16];               // per-row substep dt

  const int tid   = threadIdx.x;
  const int wave  = tid >> 5;
  const int lane  = tid & 31;
  const int blk   = blockIdx.x;            // 32 blocks x 16 batch rows
  const int lcol  = lane & 15;
  const int mrow  = (lane >> 4) << 3;      // +r gives C-layout row
  const int erow  = tid >> 4;              // elementwise: row owned by thread
  const int ecol  = (tid & 15) << 5;       // elementwise: 32-col run start

  for (int i = tid; i < 16 * YSTRIDE; i += 256) ys[i] = 0.f;
  for (int i = tid; i < 16 * ASTRIDE; i += 256) Ab[i] = (__bf16)0.f;
  __syncthreads();

  for (int j = 0; j < 32; ++j) {
    // ---- load x_t (reversed time) into Ab cols [512,768) as bf16 ----
    const int sidx = 31 - j;
    {
      const int c0 = (tid & 15) << 4;      // 16 contiguous floats per thread
      const float* xp = &x[((blk * 16 + erow) * 32 + sidx) * 256 + c0];
      __bf16* arow = &Ab[erow * ASTRIDE + 512 + c0];
      #pragma unroll
      for (int c = 0; c < 16; ++c) arow[c] = (__bf16)xp[c];
    }
    if (j > 0 && tid < 16) {
      int b = blk * 16 + tid;
      dtl[tid] = (t[b * 32 + (31 - j)] - t[b * 32 + (32 - j)]) * 0.25f;
    }
    __syncthreads();

    // ---- DOPRI5 integration: 4 substeps, 6 stages each ----
    if (j > 0) {
      for (int ss = 0; ss < 4; ++ss) {
        for (int s = 0; s < 6; ++s) {
          if (s > 0) {
            // build y_s = y + dt * sum(A[s][m] * k_m) into Ab[:,0:512)
            const float dtm = dtl[erow];
            const float* yrow = &ys[erow * YSTRIDE + ecol];
            __bf16* arow = &Ab[erow * ASTRIDE + ecol];
            #pragma unroll
            for (int cc = 0; cc < 4; ++cc) {   // 4 chunks of 8 cols
              float acc[8];
              #pragma unroll
              for (int e = 0; e < 8; ++e) acc[e] = 0.f;
              for (int m = 0; m < s && m < 5; ++m) {
                const __bf16* kp = &kb[m][erow * KSTRIDE + ecol + cc * 8];
                float da = DA[s][m];
                #pragma unroll
                for (int e = 0; e < 8; ++e) acc[e] += da * (float)kp[e];
              }
              #pragma unroll
              for (int e = 0; e < 8; ++e)
                arow[cc * 8 + e] =
                    (__bf16)(yrow[cc * 8 + e] + dtm * acc[e]);
            }
            __syncthreads();
          }
          // GEMM1: z = relu(y_s @ W1^T + b1) -> Zb. Biases issued first so
          // their latency hides behind the 64 WMMAs.
          {
            float bc[4];
            #pragma unroll
            for (int nt = 0; nt < 4; ++nt) bc[nt] = b1[wave * 64 + nt * 16 + lcol];
            v8f acc[4];
            gemm4<16>(Ab, ASTRIDE, 0, W1f, wave * 4, lane, acc);
            #pragma unroll
            for (int nt = 0; nt < 4; ++nt) {
              int col = wave * 64 + nt * 16 + lcol;
              #pragma unroll
              for (int r = 0; r < 8; ++r) {
                float v = acc[nt][r] + bc[nt];
                Zb[(r + mrow) * KSTRIDE + col] = (__bf16)(v > 0.f ? v : 0.f);
              }
            }
          }
          __syncthreads();
          // GEMM2: k_s = z @ W2^T + b2.  s<5 -> store kb[s];
          // s==5 -> fused final combine: y += dt*(B1k1+B3k3+B4k4+B5k5+B6k6)
          {
            float bc[4];
            #pragma unroll
            for (int nt = 0; nt < 4; ++nt) bc[nt] = b2[wave * 64 + nt * 16 + lcol];
            v8f acc[4];
            gemm4<16>(Zb, KSTRIDE, 0, W2f, wave * 4, lane, acc);
            if (s < 5) {
              #pragma unroll
              for (int nt = 0; nt < 4; ++nt) {
                int col = wave * 64 + nt * 16 + lcol;
                #pragma unroll
                for (int r = 0; r < 8; ++r)
                  kb[s][(r + mrow) * KSTRIDE + col] = (__bf16)(acc[nt][r] + bc[nt]);
              }
            } else {
              #pragma unroll
              for (int nt = 0; nt < 4; ++nt) {
                int col = wave * 64 + nt * 16 + lcol;
                #pragma unroll
                for (int r = 0; r < 8; ++r) {
                  int M = r + mrow;
                  int o = M * KSTRIDE + col;
                  float k6   = acc[nt][r] + bc[nt];
                  float comb = DBC[0] * (float)kb[0][o] +
                               DBC[2] * (float)kb[2][o] +
                               DBC[3] * (float)kb[3][o] +
                               DBC[4] * (float)kb[4][o] + DBC[5] * k6;
                  float ny = ys[M * YSTRIDE + col] + dtl[M] * comb;
                  ys[M * YSTRIDE + col] = ny;
                  Ab[M * ASTRIDE + col] = (__bf16)ny;  // invariant Ab[:,0:512)=y
                }
              }
            }
          }
          __syncthreads();
        }
      }
    }

    // ---- GRU cell: h = gru(x_t, y). A = [h | x] (K=768 concat in Ab) ----
    float* hstage = reinterpret_cast<float*>(&kb[0][0]);   // 16*512 f32 = 32 KB
    #pragma unroll 1
    for (int nt = 0; nt < 4; ++nt) {
      int NT  = wave * 4 + nt;
      int col = NT * 16 + lcol;
      float br = b_ih[col]        + b_hh[col];          // issue bias loads first
      float bz = b_ih[512 + col]  + b_hh[512 + col];
      float bi = b_ih[1024 + col];
      float bh = b_hh[1024 + col];
      v8f aR, aZ, aH, aI;
      gru_gates(Ab, Wrf, Wzf, Whnf, Winf, NT, lane, aR, aZ, aH, aI);
      #pragma unroll
      for (int r = 0; r < 8; ++r) {
        int M = r + mrow;
        float rg = sigmoid_hw(aR[r] + br);
        float zg = sigmoid_hw(aZ[r] + bz);
        float ng = tanh_hw(aI[r] + bi + rg * (aH[r] + bh));
        hstage[M * 512 + col] =
            (1.f - zg) * ng + zg * ys[M * YSTRIDE + col];
      }
    }
    __syncthreads();
    {
      const float* hp = &hstage[erow * 512 + ecol];
      float* yrow = &ys[erow * YSTRIDE + ecol];
      __bf16* arow = &Ab[erow * ASTRIDE + ecol];
      #pragma unroll 4
      for (int c = 0; c < 32; ++c) {
        float v = hp[c];
        yrow[c] = v;
        arow[c] = (__bf16)v;
      }
    }
    __syncthreads();
  }

  // ---- write final h (fp32) ----
  {
    const float* yrow = &ys[erow * YSTRIDE + ecol];
    float* op = &out[(blk * 16 + erow) * 512 + ecol];
    #pragma unroll 4
    for (int c = 0; c < 32; ++c) op[c] = yrow[c];
  }
}

// ---------------------------------------------------------------------------
// Entry point
// ---------------------------------------------------------------------------
extern "C" void kernel_launch(void* const* d_in, const int* in_sizes, int n_in,
                              void* d_out, int out_size, void* d_ws, size_t ws_size,
                              hipStream_t stream) {
  (void)in_sizes; (void)n_in; (void)out_size; (void)ws_size;

  const float* t    = (const float*)d_in[0];
  const float* x    = (const float*)d_in[1];
  const float* W_ih = (const float*)d_in[2];
  const float* W_hh = (const float*)d_in[3];
  const float* b_ih = (const float*)d_in[4];
  const float* b_hh = (const float*)d_in[5];
  const float* W1   = (const float*)d_in[6];
  const float* b1   = (const float*)d_in[7];
  const float* W2   = (const float*)d_in[8];
  const float* b2   = (const float*)d_in[9];
  float* out = (float*)d_out;

  unsigned short* ws   = (unsigned short*)d_ws;
  unsigned short* W1f  = ws;                 // 32 * 16 * 512 bf16
  unsigned short* W2f  = W1f  + 262144;      // 32 * 16 * 512
  unsigned short* Wrf  = W2f  + 262144;      // 32 * 24 * 512
  unsigned short* Wzf  = Wrf  + 393216;      // 32 * 24 * 512
  unsigned short* Winf = Wzf  + 393216;      // 32 *  8 * 512
  unsigned short* Whnf = Winf + 131072;      // 32 * 16 * 512

  pack_plain <<<dim3(32 * 16), dim3(32), 0, stream>>>(W1,   512,    0, 16, W1f);
  pack_plain <<<dim3(32 * 16), dim3(32), 0, stream>>>(W2,   512,    0, 16, W2f);
  pack_concat<<<dim3(32 * 24), dim3(32), 0, stream>>>(W_hh, W_ih,   0,     Wrf);
  pack_concat<<<dim3(32 * 24), dim3(32), 0, stream>>>(W_hh, W_ih, 512,     Wzf);
  pack_plain <<<dim3(32 *  8), dim3(32), 0, stream>>>(W_ih, 256, 1024,  8, Winf);
  pack_plain <<<dim3(32 * 16), dim3(32), 0, stream>>>(W_hh, 512, 1024, 16, Whnf);

  node_persistent<<<dim3(32), dim3(256), 0, stream>>>(
      t, x, b_ih, b_hh, b1, b2, W1f, W2f, Wrf, Wzf, Winf, Whnf, out);
}
